// Adaptive_Pooling_Layer_19877108646274
// MI455X (gfx1250) — matrix operand order repaired and checked by reading, not compiled, not run on hardware
//
#include <hip/hip_runtime.h>
#include <hip/hip_bf16.h>

typedef __attribute__((ext_vector_type(16))) __bf16 v16bf;
typedef __attribute__((ext_vector_type(8)))  float  v8f;
typedef unsigned short ushort_t;

#define BB 16
#define NN 2048
#define DD 256
#define HH 8
#define MM 256
#define HM (HH*MM)
#define DOUT 256

__device__ __forceinline__ ushort_t f2bf(float f) {
    unsigned u = __float_as_uint(f);
    unsigned r = u + 0x7FFFu + ((u >> 16) & 1u);   // round-to-nearest-even
    return (ushort_t)(r >> 16);
}

// CDNA5 async global->LDS copy (16B per lane), tracked by ASYNCcnt.
// ldst: generic pointer into __shared__; low 32 bits are the DS byte offset.
__device__ __forceinline__ void async_copy_b128(const void* gsrc, void* ldst) {
    unsigned lds_off = (unsigned)(unsigned long long)(uintptr_t)ldst;
    unsigned long long gaddr = (unsigned long long)(uintptr_t)gsrc;
    asm volatile("global_load_async_to_lds_b128 %0, %1, off"
                 :: "v"(lds_off), "v"(gaddr)
                 : "memory");
}

__device__ __forceinline__ void wait_async0() {
    asm volatile("s_wait_asynccnt 0x0" ::: "memory");
}

union Frag { uint4 q[2]; v16bf v; };

// ---------------- small prep kernels (fp32) ----------------

// temp[b][d] = mean_n node[b][n][d]
__global__ void mean_kernel(const float* __restrict__ node, float* __restrict__ temp) {
    int b = blockIdx.x, d = threadIdx.x;
    const float* p = node + (long long)b * NN * DD + d;
    float acc = 0.f;
    for (int n = 0; n < NN; ++n) acc += p[(long long)n * DD];
    temp[b * DD + d] = acc * (1.0f / NN);
}

// havg[b][d] = tanh(sum_k temp[b][k] * W0[k][d])
__global__ void havg_kernel(const float* __restrict__ temp, const float* __restrict__ W0,
                            float* __restrict__ havg) {
    __shared__ float t[DD];
    int b = blockIdx.x, d = threadIdx.x;
    t[d] = temp[b * DD + d];
    __syncthreads();
    float acc = 0.f;
    for (int k = 0; k < DD; ++k) acc += t[k] * W0[k * DD + d];
    havg[b * DD + d] = tanhf(acc);
}

// att[b][n] = dot(node[b][n][:], havg[b][:])   (wave32 per node, 8 nodes/block)
__global__ void att_kernel(const float* __restrict__ node, const float* __restrict__ havg,
                           float* __restrict__ att) {
    __shared__ float h[DD];
    int b = blockIdx.y;
    for (int i = threadIdx.x; i < DD; i += 256) h[i] = havg[b * DD + i];
    __syncthreads();
    int w = threadIdx.x >> 5, lane = threadIdx.x & 31;
    int n = blockIdx.x * 8 + w;
    const float* p = node + ((long long)b * NN + n) * DD;
    float acc = 0.f;
    for (int i = lane; i < DD; i += 32) acc += p[i] * h[i];
    for (int off = 16; off > 0; off >>= 1) acc += __shfl_down(acc, off, 32);
    if (lane == 0) att[(long long)b * NN + n] = acc;
}

// bc[b][d] = sum_n att[b][n] * node[b][n][d]
__global__ void bc_kernel(const float* __restrict__ node, const float* __restrict__ att,
                          float* __restrict__ bc) {
    int b = blockIdx.x, d = threadIdx.x;
    const float* p = node + (long long)b * NN * DD + d;
    const float* a = att + (long long)b * NN;
    float acc = 0.f;
    for (int n = 0; n < NN; ++n) acc += a[n] * p[(long long)n * DD];
    bc[b * DD + d] = acc;
}

// ns_bf[b][n][d] = bf16( node / max(||node||,1e-6) )   (wave32 per node)
__global__ void nodenorm_kernel(const float* __restrict__ node, ushort_t* __restrict__ nsbf) {
    int b = blockIdx.y;
    int w = threadIdx.x >> 5, lane = threadIdx.x & 31;
    int n = blockIdx.x * 8 + w;
    const float* p = node + ((long long)b * NN + n) * DD;
    float v[8];
    float ss = 0.f;
#pragma unroll
    for (int i = 0; i < 8; ++i) { v[i] = p[lane + 32 * i]; ss += v[i] * v[i]; }
    for (int off = 16; off > 0; off >>= 1) ss += __shfl_xor(ss, off, 32);
    float inv = 1.0f / fmaxf(sqrtf(ss), 1e-6f);
    ushort_t* q = nsbf + ((long long)b * NN + n) * DD;
#pragma unroll
    for (int i = 0; i < 8; ++i) q[lane + 32 * i] = f2bf(v[i] * inv);
}

// centroids MLP + L2 normalize over d -> bf16 c_n[b][hm][d]
__global__ void centroid_kernel(const float* __restrict__ bc, const float* __restrict__ i2cw,
                                const float* __restrict__ i2cb, const float* __restrict__ linw,
                                const float* __restrict__ linb, ushort_t* __restrict__ cnbf) {
    __shared__ float lsum[256];
    int hm = blockIdx.x, b = blockIdx.y, d = threadIdx.x;
    float bcd = bc[b * DD + d];
    float acc = linb[hm];
#pragma unroll
    for (int h = 0; h < HH; ++h) {
        float x = fmaxf(bcd * i2cw[h] + i2cb[h], 0.f);
        acc += x * linw[hm * HH + h];
    }
    acc = fmaxf(acc, 0.f);
    lsum[d] = acc * acc;
    __syncthreads();
    for (int s = 128; s > 0; s >>= 1) {
        if (d < s) lsum[d] += lsum[d + s];
        __syncthreads();
    }
    float inv = 1.0f / fmaxf(sqrtf(lsum[0]), 1e-6f);
    cnbf[((long long)b * HM + hm) * DD + d] = f2bf(acc * inv);
}

// normalizer over m: normhn[h][n] = sum_m cheads[h*M+m][n]
__global__ void norm_kernel(const float* __restrict__ cheads, float* __restrict__ normhn) {
    int n = blockIdx.x * 256 + threadIdx.x;
    int h = blockIdx.y;
    const float* p = cheads + (long long)h * MM * NN + n;
    float acc = 0.f;
    for (int m = 0; m < MM; ++m) acc += p[(long long)m * NN];
    normhn[h * NN + n] = acc;
}

// C_bf[m][n] = bf16( sum_h conv_w[h]*cheads[h*M+m][n]/(normhn[h][n]+1e-10) + conv_b )
__global__ void combine_kernel(const float* __restrict__ cheads, const float* __restrict__ normhn,
                               const float* __restrict__ convw, const float* __restrict__ convb,
                               ushort_t* __restrict__ Cbf) {
    int n = blockIdx.x * 256 + threadIdx.x;
    int m = blockIdx.y;
    float acc = convb[0];
#pragma unroll
    for (int h = 0; h < HH; ++h) {
        acc += convw[h] * cheads[((long long)(h * MM + m)) * NN + n] /
               (normhn[h * NN + n] + 1e-10f);
    }
    Cbf[(long long)m * NN + n] = f2bf(acc);
}

// ---------------- bf16 WMMA GEMM ----------------
// D[M_,N_] = A[M_,K_](bf16 row-major) x B, fp32 accumulate.
// BKMAJOR=0: B stored [N_,K_] (i.e. B^T row-major). BKMAJOR=1: B stored [K_,N_] (transposed in LDS).
// BF32: B source is fp32, converted to bf16 while staging to LDS.
// 128x128 block tile, 8 waves (4x2), wave tile 32x64 = 2x4 WMMA accumulators, K-tile 32.
// bf16-sourced tiles staged with GLOBAL_LOAD_ASYNC_TO_LDS_B128 (ASYNCcnt).
template <bool BKMAJOR, bool BF32, bool RELU>
__global__ __launch_bounds__(256) void gemm_wmma(
    const ushort_t* __restrict__ A, const void* __restrict__ Bp,
    float* __restrict__ outF, ushort_t* __restrict__ outB,
    const float* __restrict__ bias,
    int M_, int N_, int K_,
    long long sA, long long sB, long long sO) {
    __shared__ __align__(16) ushort_t As[128][40];
    __shared__ __align__(16) ushort_t Bs[128][40];

    const int tid = threadIdx.x;
    const int bm = blockIdx.y * 128, bn = blockIdx.x * 128;
    const int z = blockIdx.z;
    const ushort_t* Ab = A + (long long)z * sA;
    const ushort_t* Bb16 = BF32 ? nullptr : (const ushort_t*)Bp + (long long)z * sB;
    const float* Bb32 = BF32 ? (const float*)Bp + (long long)z * sB : nullptr;
    const int ldb = BKMAJOR ? N_ : K_;

    const int w = tid >> 5, lane = tid & 31;
    const int wr = w & 3, wc = w >> 2;
    const int fr = lane & 15, fg = lane >> 4;

    v8f acc[2][4] = {};

    for (int kt = 0; kt < K_; kt += 32) {
        // stage A tile: 128 rows x 32 k, async global->LDS (bf16 direct copy)
        for (int i = tid; i < 512; i += 256) {
            int r = i >> 2, s = i & 3;
            async_copy_b128(Ab + (long long)(bm + r) * K_ + kt + s * 8, &As[r][s * 8]);
        }
        // stage B tile into Bs[n][k]
        if (!BKMAJOR) {
            for (int i = tid; i < 512; i += 256) {
                int r = i >> 2, s = i & 3;
                if (BF32) {
                    const float* src = Bb32 + (long long)(bn + r) * ldb + kt + s * 8;
                    alignas(16) ushort_t tmp[8];
#pragma unroll
                    for (int j = 0; j < 8; ++j) tmp[j] = f2bf(src[j]);
                    *(uint4*)&Bs[r][s * 8] = *(const uint4*)tmp;
                } else {
                    async_copy_b128(Bb16 + (long long)(bn + r) * ldb + kt + s * 8,
                                    &Bs[r][s * 8]);
                }
            }
        } else {
            for (int i = tid; i < 512; i += 256) {
                int k = i >> 4, s = i & 15;
                int n0 = s * 8;
                if (BF32) {
                    const float* src = Bb32 + (long long)(kt + k) * ldb + bn + n0;
#pragma unroll
                    for (int j = 0; j < 8; ++j) Bs[n0 + j][k] = f2bf(src[j]);
                } else {
                    const ushort_t* src = Bb16 + (long long)(kt + k) * ldb + bn + n0;
#pragma unroll
                    for (int j = 0; j < 8; ++j) Bs[n0 + j][k] = src[j];
                }
            }
        }
        wait_async0();          // drain this wave's async LDS writes
        __syncthreads();        // publish tile across all 8 waves

        Frag a[2], b[4];
#pragma unroll
        for (int mi = 0; mi < 2; ++mi) {
            int m = wr * 32 + mi * 16 + fr;
            // A frag: lane fr holds row m; K = fg*8 + {0..7}, fg*8+16 + {0..7}
            a[mi].q[0] = *(const uint4*)&As[m][fg * 8];
            a[mi].q[1] = *(const uint4*)&As[m][fg * 8 + 16];
        }
#pragma unroll
        for (int ni = 0; ni < 4; ++ni) {
            int n = wc * 64 + ni * 16 + fr;
            // B frag: lane fr holds col n; K = fg*8 + {0..7}, fg*8+16 + {0..7}
            b[ni].q[0] = *(const uint4*)&Bs[n][fg * 8];
            b[ni].q[1] = *(const uint4*)&Bs[n][fg * 8 + 16];
        }
#pragma unroll
        for (int mi = 0; mi < 2; ++mi)
#pragma unroll
            for (int ni = 0; ni < 4; ++ni)
                acc[mi][ni] = __builtin_amdgcn_wmma_f32_16x16x32_bf16(
                    false, a[mi].v, false, b[ni].v, (short)0, acc[mi][ni], false, false);
        __syncthreads();
    }

    // epilogue: C/D layout — lane fr = column, rows fg*8 + r
#pragma unroll
    for (int mi = 0; mi < 2; ++mi) {
#pragma unroll
        for (int ni = 0; ni < 4; ++ni) {
            int n = bn + wc * 64 + ni * 16 + fr;
            float bv = bias ? bias[n] : 0.f;
#pragma unroll
            for (int r = 0; r < 8; ++r) {
                int m = bm + wr * 32 + mi * 16 + fg * 8 + r;
                float v = acc[mi][ni][r] + bv;
                if (RELU) v = fmaxf(v, 0.f);
                long long idx = (long long)z * sO + (long long)m * N_ + n;
                if (outF) outF[idx] = v;
                if (outB) outB[idx] = f2bf(v);
            }
        }
    }
}

extern "C" void kernel_launch(void* const* d_in, const int* in_sizes, int n_in,
                              void* d_out, int out_size, void* d_ws, size_t ws_size,
                              hipStream_t stream) {
    const float* node  = (const float*)d_in[0];
    const float* adj   = (const float*)d_in[1];
    const float* W0    = (const float*)d_in[2];
    const float* i2cw  = (const float*)d_in[3];
    const float* i2cb  = (const float*)d_in[4];
    const float* linw  = (const float*)d_in[5];
    const float* linb  = (const float*)d_in[6];
    const float* convw = (const float*)d_in[7];
    const float* convb = (const float*)d_in[8];
    const float* featw = (const float*)d_in[9];
    const float* featb = (const float*)d_in[10];
    float* out = (float*)d_out;

    char* wp = (char*)d_ws;
    auto alloc = [&](size_t bytes) {
        char* p = wp;
        wp += (bytes + 255) & ~(size_t)255;
        return p;
    };
    float*    temp   = (float*)alloc((size_t)BB * DD * 4);
    float*    havg   = (float*)alloc((size_t)BB * DD * 4);
    float*    att    = (float*)alloc((size_t)BB * NN * 4);
    float*    bc     = (float*)alloc((size_t)BB * DD * 4);
    ushort_t* nsbf   = (ushort_t*)alloc((size_t)BB * NN * DD * 2);
    ushort_t* cnbf   = (ushort_t*)alloc((size_t)BB * HM * DD * 2);
    float*    cheads = (float*)alloc((size_t)HM * NN * 4);       // reused per batch
    float*    normhn = (float*)alloc((size_t)HH * NN * 4);
    ushort_t* Cbf    = (ushort_t*)alloc((size_t)BB * MM * NN * 2);
    ushort_t* nnbf   = (ushort_t*)alloc((size_t)BB * MM * DD * 2);
    ushort_t* qbf    = (ushort_t*)alloc((size_t)BB * MM * NN * 2);

    // attention pooling
    mean_kernel<<<BB, 256, 0, stream>>>(node, temp);
    havg_kernel<<<BB, 256, 0, stream>>>(temp, W0, havg);
    att_kernel<<<dim3(NN / 8, BB), 256, 0, stream>>>(node, havg, att);
    bc_kernel<<<BB, 256, 0, stream>>>(node, att, bc);
    // normalized operands
    nodenorm_kernel<<<dim3(NN / 8, BB), 256, 0, stream>>>(node, nsbf);
    centroid_kernel<<<dim3(HM, BB), 256, 0, stream>>>(bc, i2cw, i2cb, linw, linb, cnbf);

    // per-batch: C_heads GEMM -> normalizer -> head-combine (reuses cheads buffer)
    for (int b = 0; b < BB; ++b) {
        gemm_wmma<false, false, false><<<dim3(16, 16, 1), 256, 0, stream>>>(
            cnbf + (long long)b * HM * DD, (const void*)(nsbf + (long long)b * NN * DD),
            cheads, nullptr, nullptr, HM, NN, DD, 0, 0, 0);
        norm_kernel<<<dim3(NN / 256, HH), 256, 0, stream>>>(cheads, normhn);
        combine_kernel<<<dim3(NN / 256, MM), 256, 0, stream>>>(
            cheads, normhn, convw, convb, Cbf + (long long)b * MM * NN);
    }

    // new_node_raw[b] = C[b] @ node[b]   (B = node fp32, K-major, converted in LDS)
    gemm_wmma<true, true, false><<<dim3(DD / 128, MM / 128, BB), 256, 0, stream>>>(
        Cbf, (const void*)node, nullptr, nnbf, nullptr,
        MM, DD, NN, (long long)MM * NN, (long long)NN * DD, (long long)MM * DD);

    // new_node = new_node_raw @ feat_w^T + feat_b  -> d_out[0 : B*M*DO]
    gemm_wmma<false, true, false><<<dim3(DOUT / 128, MM / 128, BB), 256, 0, stream>>>(
        nnbf, (const void*)featw, out, nullptr, featb,
        MM, DOUT, DD, (long long)MM * DD, 0, (long long)MM * DOUT);

    // q_adj[b] = C[b] @ adj[b]   (B = adj fp32, K-major, converted in LDS)
    gemm_wmma<true, true, false><<<dim3(NN / 128, MM / 128, BB), 256, 0, stream>>>(
        Cbf, (const void*)adj, nullptr, qbf, nullptr,
        MM, NN, NN, (long long)MM * NN, (long long)NN * NN, (long long)MM * NN);

    // new_adj[b] = relu(q_adj[b] @ C[b]^T)  -> d_out[B*M*DO : ]
    gemm_wmma<false, false, true><<<dim3(MM / 128, MM / 128, BB), 256, 0, stream>>>(
        qbf, (const void*)Cbf, out + (long long)BB * MM * DOUT, nullptr, nullptr,
        MM, MM, NN, (long long)MM * NN, (long long)MM * NN, (long long)MM * MM);

    (void)in_sizes; (void)n_in; (void)out_size; (void)ws_size;
}
